// KernelSpecifiedResnetBlock_26800595927627
// MI455X (gfx1250) — matrix operand order repaired and checked by reading, not compiled, not run on hardware
//
#include <hip/hip_runtime.h>

typedef __attribute__((ext_vector_type(16))) _Float16 v16h;
typedef __attribute__((ext_vector_type(8)))  _Float16 v8h;
typedef __attribute__((ext_vector_type(8)))  float    v8f;

#define B_    16
#define C_    256
#define HW_   64
#define WT_ELEMS (16ll * 3 * 3 * 256 * 256)   // 9,437,184 halfs per weight tensor

// ---------------------------------------------------------------------------
// Pre-pass: convert f32 weights [b][co][ci][kh][kw] -> f16 [b][kh][kw][co][ci]
// ---------------------------------------------------------------------------
__global__ __launch_bounds__(256) void wt_prep(const float* __restrict__ k1,
                                               const float* __restrict__ k2,
                                               _Float16* __restrict__ wt1,
                                               _Float16* __restrict__ wt2) {
  int t = blockIdx.x * 256 + threadIdx.x;   // 0 .. 1,179,647
  int ci8 = t & 31;
  int tmp = t >> 5;
  int co  = tmp & 255;
  tmp >>= 8;                                 // (b*3+kh)*3+kw, 0..143
  int kw = tmp % 3;
  tmp /= 3;
  int kh = tmp % 3;
  int b  = tmp / 3;

  size_t src = ((size_t)(b * 256 + co) * 256 + (size_t)ci8 * 8) * 9 + kh * 3 + kw;
  size_t dst = ((size_t)((b * 3 + kh) * 3 + kw) * 256 + co) * 256 + (size_t)ci8 * 8;

  v8h a, c;
#pragma unroll
  for (int j = 0; j < 8; ++j) {
    a[j] = (_Float16)k1[src + (size_t)j * 9];
    c[j] = (_Float16)k2[src + (size_t)j * 9];
  }
  *(v8h*)&wt1[dst] = a;
  *(v8h*)&wt2[dst] = c;
}

// ---------------------------------------------------------------------------
// Implicit-GEMM 3x3 SAME conv via v_wmma_f32_16x16x32_f16.
// Block tile: 128 c_out x 128 px (two image rows). 8 waves, each 16 c_out.
// Weights staged global->LDS with async-to-LDS (ASYNCcnt) direct path.
// PASS 1: out = f16( relu(conv(x,wt)) )          (in = f32 x)
// PASS 2: out = f32( relu(xres + conv(h,wt)) )   (in = f16 h)
// ---------------------------------------------------------------------------
template <int PASS>
__global__ __launch_bounds__(256) void conv3x3(const void* __restrict__ inv,
                                               const _Float16* __restrict__ wt,
                                               const float* __restrict__ xres,
                                               void* __restrict__ outv) {
  const int y0  = blockIdx.x * 2;   // two image rows y0, y0+1
  const int cot = blockIdx.y;       // c_out tile, 0..1
  const int b   = blockIdx.z;       // sample, 0..15

  // LDS tiles, rows padded to 40 halfs (80B) for bank spread, 16B aligned.
  __shared__ alignas(16) _Float16 Bt[2 * 66 * 40];    // [rowsel][px_src+1][ci(32)]
  __shared__ alignas(16) _Float16 At[3 * 128 * 40];   // [kw][co(128)][ci(32)]

  const int tid   = threadIdx.x;
  const int lane  = tid & 31;
  const int wv    = tid >> 5;        // wave 0..7
  const int m     = lane & 15;
  const int khalf = lane >> 4;

  // zero the halo rows of both Bt surfaces (px_src = -1 and px_src = 64)
  if (tid < 16) {
    int rs    = tid >> 3;
    int row   = ((tid >> 2) & 1) ? 65 : 0;
    int chunk = tid & 3;
    v8h z;
#pragma unroll
    for (int i = 0; i < 8; ++i) z[i] = (_Float16)0.0f;
    *(v8h*)&Bt[(rs * 66 + row) * 40 + chunk * 8] = z;
  }

  v8f acc[8] = {};   // [rowsel(2)][ntile(4)]

  const _Float16* wbase = wt + (size_t)b * (3 * 3 * 256 * 256);
  const float*    in32  = (const float*)inv;
  const _Float16* in16  = (const _Float16*)inv;
  const size_t in_base  = (size_t)b * C_ * HW_ * HW_;

  for (int kh = 0; kh < 3; ++kh) {
    for (int cc = 0; cc < 8; ++cc) {          // ci chunk of 32
      __syncthreads();

      // ---- stage B tiles (transposed, halo via row offset), both rows -----
      {
        int px = tid & 63;
        int cg = tid >> 6;                    // ci group 0..3 (8 ci each)
#pragma unroll
        for (int rs = 0; rs < 2; ++rs) {
          int  ysrc = y0 + rs + kh - 1;
          bool vrow = ((unsigned)ysrc < (unsigned)HW_);
          v8h v;
#pragma unroll
          for (int j = 0; j < 8; ++j) {
            int   ci = cc * 32 + cg * 8 + j;
            float f  = 0.0f;
            if (vrow) {
              size_t idx = in_base + ((size_t)ci * HW_ + ysrc) * HW_ + px;
              f = (PASS == 2) ? (float)in16[idx] : in32[idx];
            }
            v[j] = (_Float16)f;
          }
          *(v8h*)&Bt[(rs * 66 + px + 1) * 40 + cg * 8] = v;
        }
      }

      // ---- stage A tiles: direct global->LDS async copies (ASYNCcnt) ------
      {
#pragma unroll
        for (int j = 0; j < 6; ++j) {
          int idx   = tid + 256 * j;          // 0..1535 chunk ids
          int chunk = idx & 3;
          int row   = idx >> 2;               // 0..383
          int kw    = row >> 7;
          int col   = row & 127;
          const _Float16* src = wbase + (size_t)(kh * 3 + kw) * 65536
                              + (size_t)(cot * 128 + col) * 256 + cc * 32 + chunk * 8;
          // low 32 bits of generic shared address == LDS byte address
          unsigned ldst = (unsigned)(size_t)&At[(kw * 128 + col) * 40 + chunk * 8];
          asm volatile("global_load_async_to_lds_b128 %0, %1, off"
                       :: "v"(ldst), "v"(src) : "memory");
        }
        asm volatile("s_wait_asynccnt 0x0" ::: "memory");
      }
      __syncthreads();

      // ---- compute: 3 kw x 8 subtiles of v_wmma_f32_16x16x32_f16 ----------
#pragma unroll
      for (int kw = 0; kw < 3; ++kw) {
        // A fragment: lane m = co row, VGPR0-3 <- K[khalf*8..+8), VGPR4-7 <- K[16+khalf*8..+8)
        const _Float16* ap = &At[(kw * 128 + wv * 16 + m) * 40 + khalf * 8];
        v8h alo = *(const v8h*)ap;
        v8h ahi = *(const v8h*)(ap + 16);
        v16h a = __builtin_shufflevector(alo, ahi, 0, 1, 2, 3, 4, 5, 6, 7,
                                                   8, 9, 10, 11, 12, 13, 14, 15);
#pragma unroll
        for (int rt = 0; rt < 8; ++rt) {
          int rs = rt >> 2;                   // image row select
          int nt = rt & 3;                    // 16-px subtile
          // B fragment: lane n = pixel column, VGPR0-3 <- K[khalf*8..+8), VGPR4-7 <- K[16+khalf*8..+8)
          const _Float16* bp = &Bt[(rs * 66 + nt * 16 + m + kw) * 40 + khalf * 8];
          v8h blo = *(const v8h*)bp;
          v8h bhi = *(const v8h*)(bp + 16);
          v16h bm = __builtin_shufflevector(blo, bhi, 0, 1, 2, 3, 4, 5, 6, 7,
                                                      8, 9, 10, 11, 12, 13, 14, 15);
          acc[rt] = __builtin_amdgcn_wmma_f32_16x16x32_f16(
              false, a, false, bm, (short)0, acc[rt], false, false);
        }
      }
    }
  }

  // ---- epilogue: C/D layout -> lane n = col, VGPR r = row (lo half M=r, hi M=8+r)
  const int n    = lane & 15;
  const int mrow = (lane >> 4) * 8;
#pragma unroll
  for (int rt = 0; rt < 8; ++rt) {
    int rs = rt >> 2;
    int nt = rt & 3;
    int y  = y0 + rs;
    int px = nt * 16 + n;
#pragma unroll
    for (int r = 0; r < 8; ++r) {
      int    co   = cot * 128 + wv * 16 + mrow + r;
      size_t oidx = ((size_t)b * C_ + co) * (HW_ * HW_) + (size_t)y * HW_ + px;
      float  v    = acc[rt][r];
      if (PASS == 1) {
        v = v > 0.0f ? v : 0.0f;
        ((_Float16*)outv)[oidx] = (_Float16)v;
      } else {
        v += xres[oidx];
        v = v > 0.0f ? v : 0.0f;
        ((float*)outv)[oidx] = v;
      }
    }
  }
}

extern "C" void kernel_launch(void* const* d_in, const int* in_sizes, int n_in,
                              void* d_out, int out_size, void* d_ws, size_t ws_size,
                              hipStream_t stream) {
  const float* x  = (const float*)d_in[0];
  const float* k1 = (const float*)d_in[1];
  const float* k2 = (const float*)d_in[2];
  float* out = (float*)d_out;

  _Float16* wt1 = (_Float16*)d_ws;            // 18.87 MB
  _Float16* wt2 = wt1 + WT_ELEMS;             // 18.87 MB
  _Float16* h   = wt2 + WT_ELEMS;             // 33.55 MB (relu(conv1) intermediate)

  // 1) weight convert/transpose to f16 [b][kh][kw][co][ci]
  wt_prep<<<4608, 256, 0, stream>>>(k1, k2, wt1, wt2);

  // 2) conv1 + relu -> h (f16)
  conv3x3<1><<<dim3(HW_ / 2, 2, B_), 256, 0, stream>>>(x, wt1, nullptr, h);

  // 3) conv2 + residual + relu -> out (f32)
  conv3x3<2><<<dim3(HW_ / 2, 2, B_), 256, 0, stream>>>(h, wt2, x, out);
}
